// MultiheadAttention_13958643712197
// MI455X (gfx1250) — matrix operand order repaired and checked by reading, not compiled, not run on hardware
//
#include <hip/hip_runtime.h>
#include <hip/hip_bf16.h>
#include <stdint.h>

// ---------------------------------------------------------------------------
// MultiheadAttention forward for MI455X (gfx1250, wave32, WMMA f16->f32)
// ---------------------------------------------------------------------------
#define EMBED  1024
#define HEADS  16
#define HDIM   64
#define BSZ    8
#define SEQ    1024
#define MROWS  (BSZ * SEQ)          // 8192 rows in all projection GEMMs
#define BK     32                   // GEMM K-step
#define NKIT   (EMBED / BK)         // 32 K iterations

typedef _Float16 f16;
typedef __attribute__((ext_vector_type(16))) _Float16 v16h;
typedef __attribute__((ext_vector_type(8)))  _Float16 v8h;
typedef __attribute__((ext_vector_type(8)))  float    v8f;
typedef __attribute__((ext_vector_type(4)))  int      v4i;

#if defined(__HIP_DEVICE_COMPILE__) && \
    __has_builtin(__builtin_amdgcn_global_load_async_to_lds_b128)
#define USE_ASYNC 1
#else
#define USE_ASYNC 0
#endif

#if USE_ASYNC
#define NBUF 3
#else
#define NBUF 2
#endif

__device__ __forceinline__ v8f wmma_f16(v16h a, v16h b, v8f c) {
  // D = A(16x32 f16) * B(32x16 f16) + C(16x16 f32)
  return __builtin_amdgcn_wmma_f32_16x16x32_f16(
      /*neg_a=*/false, a, /*neg_b=*/false, b,
      /*c_mod=*/(short)0, c, /*reuse_a=*/false, /*reuse_b=*/false);
}

__device__ __forceinline__ v16h pack16(v8h lo, v8h hi) {
  v16h r;
#pragma unroll
  for (int i = 0; i < 8; ++i) { r[i] = lo[i]; r[i + 8] = hi[i]; }
  return r;
}

#if USE_ASYNC
// async copy of 16 bytes global -> LDS (ASYNCcnt-tracked, bypasses VGPRs)
// prototype (from clang diagnostic): (v4i addrspace(1)*, v4i addrspace(3)*,
//                                     imm int offset, imm int cpol)
__device__ __forceinline__ void async_b128(const f16* g, f16* l) {
  __builtin_amdgcn_global_load_async_to_lds_b128(
      (__attribute__((address_space(1))) v4i*)(uintptr_t)g,
      (__attribute__((address_space(3))) v4i*)(uint32_t)(uintptr_t)l,
      0, 0);
}
__device__ __forceinline__ void wait_async4() {
#if __has_builtin(__builtin_amdgcn_s_wait_asynccnt)
  __builtin_amdgcn_s_wait_asynccnt(4);
#else
  asm volatile("s_wait_asynccnt 0x4" ::: "memory");
#endif
}
__device__ __forceinline__ void wait_async0() {
#if __has_builtin(__builtin_amdgcn_s_wait_asynccnt)
  __builtin_amdgcn_s_wait_asynccnt(0);
#else
  asm volatile("s_wait_asynccnt 0x0" ::: "memory");
#endif
}
#endif

// ---------------------------------------------------------------------------
// fp32 -> f16 convert, 8 elements/thread (two float4 loads, one b128 store)
// ---------------------------------------------------------------------------
__global__ void cvt_f32_f16(const float* __restrict__ src,
                            f16* __restrict__ dst, int n8) {
  int i = blockIdx.x * blockDim.x + threadIdx.x;
  const int stride = gridDim.x * blockDim.x;
  for (; i < n8; i += stride) {
    const float4* s = (const float4*)src + (size_t)i * 2;
    const float4 a = s[0], b = s[1];
    v8h r;
    r[0] = (f16)a.x; r[1] = (f16)a.y; r[2] = (f16)a.z; r[3] = (f16)a.w;
    r[4] = (f16)b.x; r[5] = (f16)b.y; r[6] = (f16)b.z; r[7] = (f16)b.w;
    *(v8h*)(dst + (size_t)i * 8) = r;
  }
}

// ---------------------------------------------------------------------------
// Tiled WMMA GEMM: out = (A[M,K] * W[N,K]^T + bias[N]) * scale
//  mode 0: outF[M,N] fp32 row-major       (final projection -> d_out)
//  mode 1: outH as [B, H, S, D] f16       (Q, K)
//  mode 2: outH as [B, H, D, S] f16       (V transposed for the PV matmul)
// Block: 256 threads (8 waves, 4(M) x 2(N)), tile 128x128, K-step 32.
// Staging: async global->LDS 3-buffer pipeline (1 barrier/iter), or
// register-staged double buffer if the async builtin is unavailable.
// ---------------------------------------------------------------------------
__global__ __launch_bounds__(256)
void gemm_proj(const f16* __restrict__ A, const f16* __restrict__ W,
               const float* __restrict__ bias,
               f16* __restrict__ outH, float* __restrict__ outF,
               float scale, int mode) {
  const int K = EMBED, N = EMBED;
  __shared__ f16 As[NBUF][128][40];   // +8 halves pad for conflict-free b128
  __shared__ f16 Bs[NBUF][128][40];

  const int tid  = threadIdx.x;
  const int lane = tid & 31;
  const int wv   = tid >> 5;
  const int wm   = wv & 3;        // wave row  (4 waves along M)
  const int wn   = wv >> 2;       // wave col  (2 waves along N)
  const int l16  = lane & 15;
  const int hlf  = lane >> 4;     // 0 | 1  (lane half)

  const int blockRow = blockIdx.y * 128;
  const int blockCol = blockIdx.x * 128;

  v8f acc[2][4];
#pragma unroll
  for (int mi = 0; mi < 2; ++mi)
#pragma unroll
    for (int nj = 0; nj < 4; ++nj) acc[mi][nj] = (v8f){};

  // staging: 256 threads x 16 halves = 128x32 tile per matrix
  const int srow = tid >> 1;
  const int scol = (tid & 1) * 16;
  const f16* agBase = A + (size_t)(blockRow + srow) * K + scol;
  const f16* wgBase = W + (size_t)(blockCol + srow) * K + scol;

  auto compute = [&](int b) {
    v16h af[2], bf[4];
#pragma unroll
    for (int mi = 0; mi < 2; ++mi) {
      const int r  = wm * 32 + mi * 16 + l16;
      const int c0 = hlf * 16;
      af[mi] = pack16(*(const v8h*)&As[b][r][c0],
                      *(const v8h*)&As[b][r][c0 + 8]);
    }
#pragma unroll
    for (int nj = 0; nj < 4; ++nj) {
      const int r  = wn * 64 + nj * 16 + l16;
      const int c0 = hlf * 16;
      bf[nj] = pack16(*(const v8h*)&Bs[b][r][c0],
                      *(const v8h*)&Bs[b][r][c0 + 8]);
    }
#pragma unroll
    for (int mi = 0; mi < 2; ++mi)
#pragma unroll
      for (int nj = 0; nj < 4; ++nj)
        acc[mi][nj] = wmma_f16(af[mi], bf[nj], acc[mi][nj]);
  };

#if USE_ASYNC
  auto copy_tile = [&](int buf, int kb) {
#pragma unroll
    for (int p = 0; p < 2; ++p) {
      async_b128(agBase + kb + p * 8, &As[buf][srow][scol + p * 8]);
      async_b128(wgBase + kb + p * 8, &Bs[buf][srow][scol + p * 8]);
    }
  };
  copy_tile(0, 0);
  int cur = 0, nxt = 1;
  for (int it = 0; it < NKIT; ++it) {
    if (it + 1 < NKIT) {
      copy_tile(nxt, (it + 1) * BK);
      wait_async4();                 // in-order: tile `it` now resident
    } else {
      wait_async0();
    }
    __syncthreads();                 // all waves' copies for `cur` visible;
                                     // 3 buffers make 1 barrier/iter race-free
    compute(cur);
    cur = nxt;
    nxt = (nxt + 1 == NBUF) ? 0 : nxt + 1;
  }
#else
  v8h ra[2], rb[2];
  auto load_regs = [&](int kb) {
#pragma unroll
    for (int p = 0; p < 2; ++p) {
      ra[p] = *(const v8h*)(agBase + kb + p * 8);
      rb[p] = *(const v8h*)(wgBase + kb + p * 8);
    }
  };
  auto store_regs = [&](int buf) {
#pragma unroll
    for (int p = 0; p < 2; ++p) {
      *(v8h*)&As[buf][srow][scol + p * 8] = ra[p];
      *(v8h*)&Bs[buf][srow][scol + p * 8] = rb[p];
    }
  };
  load_regs(0);
  store_regs(0);
  __syncthreads();
  for (int it = 0; it < NKIT; ++it) {
    if (it + 1 < NKIT) load_regs((it + 1) * BK);   // prefetch overlaps WMMA
    compute(it & 1);
    if (it + 1 < NKIT) store_regs((it + 1) & 1);
    __syncthreads();
  }
#endif

  // Epilogue. C layout: row = v + 8*hlf (+tile), col = l16 (+tile)
#pragma unroll
  for (int mi = 0; mi < 2; ++mi)
#pragma unroll
    for (int nj = 0; nj < 4; ++nj)
#pragma unroll
      for (int v = 0; v < 8; ++v) {
        const int row = blockRow + wm * 32 + mi * 16 + v + hlf * 8;
        const int col = blockCol + wn * 64 + nj * 16 + l16;
        const float val = (acc[mi][nj][v] + bias[col]) * scale;
        if (mode == 0) {
          outF[(size_t)row * N + col] = val;
        } else {
          const int bb = row >> 10, s = row & (SEQ - 1);
          const int h  = col >> 6,  d = col & (HDIM - 1);
          const size_t idx = (mode == 1)
              ? ((size_t)(bb * HEADS + h) * SEQ + s) * HDIM + d    // [B,H,S,D]
              : ((size_t)(bb * HEADS + h) * HDIM + d) * SEQ + s;   // [B,H,D,S]
          outH[idx] = (f16)val;
        }
      }
}

// ---------------------------------------------------------------------------
// Flash attention: scores = q k^T (scale folded into Q), online softmax,
// o = P v.  Grid (SEQ/256, B*H); each wave owns 32 query rows.
// Q,K: [B,H,S,D] f16.  Vt: [B,H,D,S] f16.  O: [B,S,E] f16.
// ---------------------------------------------------------------------------
__global__ __launch_bounds__(256)
void flash_attn(const f16* __restrict__ Q, const f16* __restrict__ Kh,
                const f16* __restrict__ Vt, f16* __restrict__ O) {
  __shared__ f16 pbuf[8][32][40];   // per-wave P relayout buffer

  const int tid  = threadIdx.x;
  const int lane = tid & 31;
  const int wv   = tid >> 5;
  const int l16  = lane & 15;
  const int hlf  = lane >> 4;

  const int bh = blockIdx.y;               // 0..B*H-1
  const int bb = bh >> 4, h = bh & (HEADS - 1);
  const int t0 = blockIdx.x * 256 + wv * 32;

  const f16* q  = Q  + (size_t)bh * SEQ * HDIM;
  const f16* k  = Kh + (size_t)bh * SEQ * HDIM;
  const f16* vt = Vt + (size_t)bh * HDIM * SEQ;

  // Q A-fragments, D=64 -> two K=32 slabs per 16-row tile
  v16h aq[2][2];
#pragma unroll
  for (int mi = 0; mi < 2; ++mi)
#pragma unroll
    for (int kk = 0; kk < 2; ++kk) {
      const int r  = t0 + mi * 16 + l16;
      const int c0 = kk * 32 + hlf * 16;
      aq[mi][kk] = pack16(*(const v8h*)(q + (size_t)r * HDIM + c0),
                          *(const v8h*)(q + (size_t)r * HDIM + c0 + 8));
    }

  v8f o[2][4];
#pragma unroll
  for (int mi = 0; mi < 2; ++mi)
#pragma unroll
    for (int dj = 0; dj < 4; ++dj) o[mi][dj] = (v8f){};
  float m_[2][8], l_[2][8];
#pragma unroll
  for (int mi = 0; mi < 2; ++mi)
#pragma unroll
    for (int v = 0; v < 8; ++v) { m_[mi][v] = -3.0e38f; l_[mi][v] = 0.0f; }

  for (int s0 = 0; s0 < SEQ; s0 += 32) {
    // K B-fragments: lane = column s, contiguous d halves
    v16h bk[2][2];
#pragma unroll
    for (int kk = 0; kk < 2; ++kk)
#pragma unroll
      for (int nj = 0; nj < 2; ++nj) {
        const int s  = s0 + nj * 16 + l16;
        const int d0 = kk * 32 + hlf * 16;
        bk[kk][nj] = pack16(*(const v8h*)(k + (size_t)s * HDIM + d0),
                            *(const v8h*)(k + (size_t)s * HDIM + d0 + 8));
      }
    v8f sc[2][2];
#pragma unroll
    for (int mi = 0; mi < 2; ++mi)
#pragma unroll
      for (int nj = 0; nj < 2; ++nj) {
        sc[mi][nj] = (v8f){};
#pragma unroll
        for (int kk = 0; kk < 2; ++kk)
          sc[mi][nj] = wmma_f16(aq[mi][kk], bk[kk][nj], sc[mi][nj]);
      }

    // online softmax + stash P into per-wave LDS (A-fragment relayout)
#pragma unroll
    for (int mi = 0; mi < 2; ++mi) {
#pragma unroll
      for (int v = 0; v < 8; ++v) {
        float cm = fmaxf(sc[mi][0][v], sc[mi][1][v]);
        cm = fmaxf(cm, __shfl_xor(cm, 1));
        cm = fmaxf(cm, __shfl_xor(cm, 2));
        cm = fmaxf(cm, __shfl_xor(cm, 4));
        cm = fmaxf(cm, __shfl_xor(cm, 8));
        const float mo = m_[mi][v];
        const float mn = fmaxf(mo, cm);
        const float alpha = __expf(mo - mn);
        const float p0 = __expf(sc[mi][0][v] - mn);
        const float p1 = __expf(sc[mi][1][v] - mn);
        sc[mi][0][v] = p0;
        sc[mi][1][v] = p1;
        float rs = p0 + p1;
        rs += __shfl_xor(rs, 1);
        rs += __shfl_xor(rs, 2);
        rs += __shfl_xor(rs, 4);
        rs += __shfl_xor(rs, 8);
        l_[mi][v] = l_[mi][v] * alpha + rs;
        m_[mi][v] = mn;
#pragma unroll
        for (int dj = 0; dj < 4; ++dj) o[mi][dj][v] *= alpha;
      }
#pragma unroll
      for (int nj = 0; nj < 2; ++nj)
#pragma unroll
        for (int v = 0; v < 8; ++v)
          pbuf[wv][mi * 16 + v + hlf * 8][nj * 16 + l16] = (f16)sc[mi][nj][v];
    }
    // LDS ops are in-order within a wave (DScnt); same-wave RAW is safe.
    v16h ap[2];
#pragma unroll
    for (int mi = 0; mi < 2; ++mi) {
      const int r  = mi * 16 + l16;
      const int c0 = hlf * 16;
      ap[mi] = pack16(*(const v8h*)&pbuf[wv][r][c0],
                      *(const v8h*)&pbuf[wv][r][c0 + 8]);
    }
    // V B-fragments from transposed V: lane = column d, contiguous s halves
    v16h bv[4];
#pragma unroll
    for (int dj = 0; dj < 4; ++dj) {
      const int d  = dj * 16 + l16;
      const int ss = s0 + hlf * 16;
      bv[dj] = pack16(*(const v8h*)(vt + (size_t)d * SEQ + ss),
                      *(const v8h*)(vt + (size_t)d * SEQ + ss + 8));
    }
#pragma unroll
    for (int mi = 0; mi < 2; ++mi)
#pragma unroll
      for (int dj = 0; dj < 4; ++dj)
        o[mi][dj] = wmma_f16(ap[mi], bv[dj], o[mi][dj]);
  }

  // normalize and write O as [B, S, E] f16
#pragma unroll
  for (int mi = 0; mi < 2; ++mi)
#pragma unroll
    for (int dj = 0; dj < 4; ++dj)
#pragma unroll
      for (int v = 0; v < 8; ++v) {
        const int t = t0 + mi * 16 + v + hlf * 8;
        const int e = h * HDIM + dj * 16 + l16;
        const float val = o[mi][dj][v] / l_[mi][v];
        O[((size_t)(bb * SEQ + t)) * EMBED + e] = (f16)val;
      }
}

// ---------------------------------------------------------------------------
extern "C" void kernel_launch(void* const* d_in, const int* in_sizes, int n_in,
                              void* d_out, int out_size, void* d_ws,
                              size_t ws_size, hipStream_t stream) {
  const float* query = (const float*)d_in[0];
  const float* key   = (const float*)d_in[1];
  const float* value = (const float*)d_in[2];
  const float* Wq    = (const float*)d_in[3];
  const float* bq    = (const float*)d_in[4];
  const float* Wk    = (const float*)d_in[5];
  const float* bk    = (const float*)d_in[6];
  const float* Wv    = (const float*)d_in[7];
  const float* bv    = (const float*)d_in[8];
  const float* Wo    = (const float*)d_in[9];
  const float* bo    = (const float*)d_in[10];

  const size_t XN = (size_t)MROWS * EMBED;   // 8M elems
  const size_t WN = (size_t)EMBED * EMBED;   // 1M elems
  f16* ws  = (f16*)d_ws;
  f16* Xq  = ws;                 // converted inputs
  f16* Xk  = Xq + XN;
  f16* Xv  = Xk + XN;
  f16* Wqh = Xv + XN;            // converted weights
  f16* Wkh = Wqh + WN;
  f16* Wvh = Wkh + WN;
  f16* Woh = Wvh + WN;
  f16* Qh  = Woh + WN;           // projected, [B,H,S,D]
  f16* Kh  = Qh + XN;
  f16* Vt  = Kh + XN;            // projected, [B,H,D,S]
  f16* Oh  = Vt + XN;            // attention output, [B,S,E]

  const int CB = 256;
  cvt_f32_f16<<<2048, CB, 0, stream>>>(query, Xq, (int)(XN / 8));
  cvt_f32_f16<<<2048, CB, 0, stream>>>(key,   Xk, (int)(XN / 8));
  cvt_f32_f16<<<2048, CB, 0, stream>>>(value, Xv, (int)(XN / 8));
  cvt_f32_f16<<<512,  CB, 0, stream>>>(Wq, Wqh, (int)(WN / 8));
  cvt_f32_f16<<<512,  CB, 0, stream>>>(Wk, Wkh, (int)(WN / 8));
  cvt_f32_f16<<<512,  CB, 0, stream>>>(Wv, Wvh, (int)(WN / 8));
  cvt_f32_f16<<<512,  CB, 0, stream>>>(Wo, Woh, (int)(WN / 8));

  const dim3 ggrid(EMBED / 128, MROWS / 128);   // (8, 64)
  const float scaling = 0.125f;                 // HEAD_DIM^-0.5
  gemm_proj<<<ggrid, 256, 0, stream>>>(Xq, Wqh, bq, Qh, nullptr, scaling, 1);
  gemm_proj<<<ggrid, 256, 0, stream>>>(Xk, Wkh, bk, Kh, nullptr, 1.0f, 1);
  gemm_proj<<<ggrid, 256, 0, stream>>>(Xv, Wvh, bv, Vt, nullptr, 1.0f, 2);

  flash_attn<<<dim3(SEQ / 256, BSZ * HEADS), 256, 0, stream>>>(Qh, Kh, Vt, Oh);

  gemm_proj<<<ggrid, 256, 0, stream>>>(Oh, Woh, bo, nullptr, (float*)d_out,
                                       1.0f, 0);
}
